// Net_17643725652273
// MI455X (gfx1250) — matrix operand order repaired and checked by reading, not compiled, not run on hardware
//
#include <hip/hip_runtime.h>

// ---------------------------------------------------------------------------
// Types for CDNA5 WMMA (wave32): v_wmma_f32_16x16x32_bf16
// ---------------------------------------------------------------------------
typedef __attribute__((ext_vector_type(16))) __bf16          bf16x16;
typedef __attribute__((ext_vector_type(8)))  float           v8f;
typedef __attribute__((ext_vector_type(8)))  unsigned short  ushort8;

union FragBF {
    bf16x16 v;
    ushort8 h[2];   // two 16-byte halves
};

__device__ __forceinline__ unsigned short f32_to_bf16(float f) {
    unsigned int u = __float_as_uint(f);
    unsigned int r = (u + 0x7FFFu + ((u >> 16) & 1u)) >> 16;   // RNE
    return (unsigned short)r;
}

__device__ __forceinline__ float elu1(float x) {
    return x > 0.f ? x : (__expf(x) - 1.f);
}

// ---------------------------------------------------------------------------
// Pack A (fp32 [4096x4096]) -> bf16 row-major
// ---------------------------------------------------------------------------
__global__ void pack_a_bf16(const float* __restrict__ a,
                            unsigned short* __restrict__ abf, long n) {
    long i = ((long)blockIdx.x * blockDim.x + threadIdx.x) * 4;
    if (i + 3 < n) {
        float4 f = *(const float4*)(a + i);
        ushort4 o;
        o.x = f32_to_bf16(f.x); o.y = f32_to_bf16(f.y);
        o.z = f32_to_bf16(f.z); o.w = f32_to_bf16(f.w);
        *(ushort4*)(abf + i) = o;
    }
}

// ---------------------------------------------------------------------------
// Pack X [B=32, N=4096, F=16] fp32 -> Xt bf16 [512 cols][4096 k]
//   Xt[(b*16+f)*4096 + m] = x[b][m][f]   (B-operand transposed layout)
// ---------------------------------------------------------------------------
__global__ void pack_x_t(const float* __restrict__ x,
                         unsigned short* __restrict__ xt) {
    int o = blockIdx.x * blockDim.x + threadIdx.x;   // < 512*4096
    int c = o >> 12;
    int m = o & 4095;
    int b = c >> 4, f = c & 15;
    float v = x[((long)b * 4096 + m) * 16 + f];
    xt[o] = f32_to_bf16(v);
}

// ---------------------------------------------------------------------------
// Big GEMM: C[M x ncols] = A[M x K] (bf16) * B (bf16, stored as Bt[ncols][K])
// Block: 128 threads = 4 waves (2M x 2N). Wave tile 64(M) x 64(N):
//   4 A-fragments x 4 B-fragments -> 16 v_wmma per K-step of 32,
//   16 b128 loads per K-step (load:WMMA = 1.0).
// Block tile 128M x 128N.  All operands are L2-resident.
// ---------------------------------------------------------------------------
__global__ __launch_bounds__(128) void gemm_bf16_wmma(
    const unsigned short* __restrict__ A,
    const unsigned short* __restrict__ Bt,
    float* __restrict__ C, int K, int ncols)
{
    const int lane  = threadIdx.x & 31;
    const int wave  = threadIdx.x >> 5;   // 0..3
    const int waveM = wave >> 1;          // 0..1
    const int waveN = wave & 1;           // 0..1
    const int m0 = blockIdx.y * 128 + waveM * 64;   // four 16-row tiles
    const int n0 = blockIdx.x * 128 + waveN * 64;   // four 16-col tiles
    const int l15 = lane & 15;
    const int hi  = lane >> 4;

    // A fragment (16x32 bf16): lanes 0-15 -> K=0..7 / 16..23; lanes 16-31 -> K=8..15 / 24..31
    const unsigned short* aP[4];
    // B fragment (32x16 bf16): lanes 0-15 -> K=0..7 / 16..23; lanes 16-31 -> K=8..15 / 24..31
    const unsigned short* bP[4];
#pragma unroll
    for (int i = 0; i < 4; ++i) {
        aP[i] = A  + (size_t)(m0 + i * 16 + l15) * K + hi * 8;
        bP[i] = Bt + (size_t)(n0 + i * 16 + l15) * K + hi * 8;
    }

    v8f acc[4][4];
#pragma unroll
    for (int i = 0; i < 4; ++i)
#pragma unroll
        for (int t = 0; t < 4; ++t) acc[i][t] = (v8f){};

    for (int k = 0; k < K; k += 32) {
        FragBF b[4];
#pragma unroll
        for (int t = 0; t < 4; ++t) {
            b[t].h[0] = *(const ushort8*)(bP[t] + k);
            b[t].h[1] = *(const ushort8*)(bP[t] + k + 16);
        }
#pragma unroll
        for (int i = 0; i < 4; ++i) {
            FragBF a;
            a.h[0] = *(const ushort8*)(aP[i] + k);
            a.h[1] = *(const ushort8*)(aP[i] + k + 16);
#pragma unroll
            for (int t = 0; t < 4; ++t) {
                acc[i][t] = __builtin_amdgcn_wmma_f32_16x16x32_bf16(
                    false, a.v, false, b[t].v, (short)0, acc[i][t], false, false);
            }
        }
    }

    // D layout: VGPR r -> row (r + hi*8), col = l15
#pragma unroll
    for (int i = 0; i < 4; ++i) {
        const int r0 = m0 + i * 16 + hi * 8;
#pragma unroll
        for (int r = 0; r < 8; ++r) {
            float* row = C + (size_t)(r0 + r) * ncols + n0 + l15;
#pragma unroll
            for (int t = 0; t < 4; ++t) row[t * 16] = acc[i][t][r];
        }
    }
}

// ---------------------------------------------------------------------------
// Layer-1 epilogue: h1[b,n,j] = elu(C1[n, b*16+f] . w1[f,j] + b1[j])
// stored transposed as bf16 H1t[(b*32+j)*4096 + n] for GEMM2's B operand.
// Mapping t = c*4096 + n: coalesced h1t writes; per-thread 64B C1 reads;
// wave-uniform (scalar) w1/b1 reads.
// ---------------------------------------------------------------------------
__global__ void layer1_epilogue(const float* __restrict__ C1,
                                const float* __restrict__ w1,
                                const float* __restrict__ b1,
                                unsigned short* __restrict__ h1t) {
    int t = blockIdx.x * blockDim.x + threadIdx.x;   // < 1024*4096
    int c = t >> 12;          // bj  (uniform within a block)
    int n = t & 4095;
    int b = c >> 5;
    int j = c & 31;
    const float* row = C1 + (size_t)n * 512 + b * 16;
    float h = b1[j];
#pragma unroll
    for (int f = 0; f < 16; ++f) h += row[f] * w1[f * 32 + j];
    h1t[t] = f32_to_bf16(elu1(h));
}

// ---------------------------------------------------------------------------
// Layer-2 epilogue + global sum pool:
//   h2[b,n,j] = elu(C2[n, b*32+q] . w2[q,j] + b2[j]);  pool[b,j] += h2
// One block = 1024 (b,j) threads, each accumulates 64 rows then one atomic.
// ---------------------------------------------------------------------------
__global__ __launch_bounds__(1024) void layer2_epilogue_pool(
    const float* __restrict__ C2, const float* __restrict__ w2,
    const float* __restrict__ b2, float* __restrict__ pool)
{
    int bj = threadIdx.x;             // 0..1023
    int b  = bj >> 5, j = bj & 31;
    int n0 = blockIdx.x * 64;
    float acc = 0.f;
    for (int n = n0; n < n0 + 64; ++n) {
        const float* row = C2 + (size_t)n * 1024 + b * 32;
        float h = b2[j];
#pragma unroll
        for (int q = 0; q < 32; ++q) h += row[q] * w2[q * 32 + j];
        acc += elu1(h);
    }
    atomicAdd(&pool[bj], acc);
}

__global__ void zero_pool(float* __restrict__ pool) { pool[threadIdx.x] = 0.f; }

// ---------------------------------------------------------------------------
// FC head: t1 = relu(pool @ fcw1 + fcb1)   [32 x 512]
// ---------------------------------------------------------------------------
__global__ void head_fc1(const float* __restrict__ pool,
                         const float* __restrict__ fcw1,
                         const float* __restrict__ fcb1,
                         float* __restrict__ t1) {
    int i = blockIdx.x * blockDim.x + threadIdx.x;   // < 32*512
    int b = i >> 9, k = i & 511;
    float s = fcb1[k];
#pragma unroll
    for (int j = 0; j < 32; ++j) s += pool[b * 32 + j] * fcw1[j * 512 + k];
    t1[i] = s > 0.f ? s : 0.f;
}

// ---------------------------------------------------------------------------
// FC head 2 + softmax -> out [32 x 10]
// ---------------------------------------------------------------------------
__global__ void head_fc2_softmax(const float* __restrict__ t1,
                                 const float* __restrict__ fcw2,
                                 const float* __restrict__ fcb2,
                                 float* __restrict__ out) {
    __shared__ float logits[320];
    int i = threadIdx.x;              // 0..319
    if (i < 320) {
        int b = i / 10, o = i % 10;
        float s = fcb2[o];
        for (int k = 0; k < 512; ++k) s += t1[b * 512 + k] * fcw2[k * 10 + o];
        logits[i] = s;
    }
    __syncthreads();
    if (i < 32) {
        int b = i;
        float m = -1e30f;
        for (int o = 0; o < 10; ++o) m = fmaxf(m, logits[b * 10 + o]);
        float e[10], sum = 0.f;
        for (int o = 0; o < 10; ++o) { e[o] = __expf(logits[b * 10 + o] - m); sum += e[o]; }
        float inv = 1.f / sum;
        for (int o = 0; o < 10; ++o) out[b * 10 + o] = e[o] * inv;
    }
}

// ---------------------------------------------------------------------------
// kernel_launch
// ---------------------------------------------------------------------------
extern "C" void kernel_launch(void* const* d_in, const int* in_sizes, int n_in,
                              void* d_out, int out_size, void* d_ws, size_t ws_size,
                              hipStream_t stream) {
    const float* x    = (const float*)d_in[0];   // [32,4096,16]
    const float* a    = (const float*)d_in[1];   // [4096,4096]
    const float* w1   = (const float*)d_in[2];   // [16,32]
    const float* b1   = (const float*)d_in[3];   // [32]
    const float* w2   = (const float*)d_in[4];   // [32,32]
    const float* b2   = (const float*)d_in[5];   // [32]
    const float* fcw1 = (const float*)d_in[6];   // [32,512]
    const float* fcb1 = (const float*)d_in[7];   // [512]
    const float* fcw2 = (const float*)d_in[8];   // [512,10]
    const float* fcb2 = (const float*)d_in[9];   // [10]
    float* out = (float*)d_out;                  // [32,10]
    (void)in_sizes; (void)n_in; (void)out_size; (void)ws_size;

    char* ws = (char*)d_ws;
    size_t off = 0;
    auto alloc = [&](size_t bytes) -> char* {
        char* p = ws + off;
        off = (off + bytes + 255) & ~(size_t)255;
        return p;
    };
    unsigned short* a_bf = (unsigned short*)alloc((size_t)4096 * 4096 * 2); // 32 MB
    unsigned short* xt   = (unsigned short*)alloc((size_t)512  * 4096 * 2); //  4 MB
    unsigned short* h1t  = (unsigned short*)alloc((size_t)1024 * 4096 * 2); //  8 MB
    float* C1   = (float*)alloc((size_t)4096 * 512  * 4);                   //  8 MB
    float* C2   = (float*)alloc((size_t)4096 * 1024 * 4);                   // 16 MB
    float* pool = (float*)alloc(1024 * 4);
    float* t1   = (float*)alloc(32 * 512 * 4);

    // Pack operands to bf16 WMMA layouts
    pack_a_bf16<<<16384, 256, 0, stream>>>(a, a_bf, (long)4096 * 4096);
    pack_x_t<<<(512 * 4096) / 256, 256, 0, stream>>>(x, xt);
    zero_pool<<<1, 1024, 0, stream>>>(pool);

    // Layer 1: agg1 = A @ X   (C1: [4096, 512])
    gemm_bf16_wmma<<<dim3(4, 32), 128, 0, stream>>>(a_bf, xt, C1, 4096, 512);
    layer1_epilogue<<<(4096 * 1024) / 256, 256, 0, stream>>>(C1, w1, b1, h1t);

    // Layer 2: agg2 = A @ H1  (C2: [4096, 1024])
    gemm_bf16_wmma<<<dim3(8, 32), 128, 0, stream>>>(a_bf, h1t, C2, 4096, 1024);
    layer2_epilogue_pool<<<64, 1024, 0, stream>>>(C2, w2, b2, pool);

    // FC head + softmax
    head_fc1<<<64, 256, 0, stream>>>(pool, fcw1, fcb1, t1);
    head_fc2_softmax<<<1, 320, 0, stream>>>(t1, fcw2, fcb2, out);
}
